// ReformerModel_76476187672669
// MI455X (gfx1250) — matrix operand order repaired and checked by reading, not compiled, not run on hardware
//
#include <hip/hip_runtime.h>
#include <hip/hip_bf16.h>
#include <math.h>

typedef __attribute__((ext_vector_type(16))) _Float16 v16h;
typedef __attribute__((ext_vector_type(8)))  _Float16 v8h;
typedef __attribute__((ext_vector_type(8)))  float    v8f;

#define N_B    16
#define SEQ    1000
#define NTOK   16000       /* N_B*SEQ */
#define DMODEL 128
#define NHEAD  8
#define DHEAD  16
#define NBH    128         /* N_B*NHEAD */
#define NROUND 4
#define CHUNK  25
#define NCHUNK 160         /* NROUND*SEQ/CHUNK */
#define RSLOTS 4000        /* NROUND*SEQ */
#define RSPAD  4096
#define NEGV   (-1e9f)
#define SELFV  (-5e4f)

#define FLAG_ADD  1
#define FLAG_GELU 2

__device__ __forceinline__ float gelu_exact(float x) {
  return 0.5f * x * (1.0f + erff(x * 0.70710678118654752f));
}
__device__ __forceinline__ v8f wmma16(v16h a, v16h b, v8f c) {
  return __builtin_amdgcn_wmma_f32_16x16x32_f16(false, a, false, b, (short)0, c, false, false);
}

// ---------------------------------------------------------------------------
// Weight convert: f32 [K][N] row-major -> f16 Bt [Npad][K] (transposed, zero pad)
// ---------------------------------------------------------------------------
__global__ void k_w2h(const float* __restrict__ src, _Float16* __restrict__ dst,
                      int K, int N, int Npad) {
  int i = blockIdx.x * blockDim.x + threadIdx.x;
  if (i >= Npad * K) return;
  int n = i / K, k = i % K;
  dst[i] = (n < N) ? (_Float16)src[(size_t)k * N + n] : (_Float16)0.f;
}

// ---------------------------------------------------------------------------
// Embedding concat + sinusoidal positional encoding + keep mask
// ---------------------------------------------------------------------------
__global__ void k_embed(const int* __restrict__ xm, const int* __restrict__ xa,
                        const int* __restrict__ xh, const int* __restrict__ xw,
                        const float* __restrict__ em, const float* __restrict__ ea,
                        const float* __restrict__ eh, const float* __restrict__ ew,
                        float* __restrict__ x1, float* __restrict__ x2,
                        int* __restrict__ keep) {
  int i = blockIdx.x * blockDim.x + threadIdx.x;
  if (i >= NTOK) return;
  int s = i % SEQ;
  int im = xm[i], ia = xa[i], ih = xh[i], iw = xw[i];
  keep[i] = (im != 0) ? 1 : 0;
  float* p1 = x1 + (size_t)i * DMODEL;
  float* p2 = x2 + (size_t)i * DMODEL;
  for (int d = 0; d < DMODEL; ++d) {
    float e;
    if (d < 64)       e = em[im * 64 + d];
    else if (d < 96)  e = ea[ia * 32 + (d - 64)];
    else if (d < 112) e = eh[ih * 16 + (d - 96)];
    else              e = ew[iw * 16 + (d - 112)];
    int i2 = d >> 1;
    float div = expf((float)(2 * i2) * (-9.21034037198f / 128.0f));
    float ang = (float)s * div;
    float pe = (d & 1) ? cosf(ang) : sinf(ang);
    float vo = e + pe;
    p1[d] = vo; p2[d] = vo;
  }
}

// ---------------------------------------------------------------------------
// LayerNorm over 128 dims, one wave32 per row (4 rows per block); optional
// second operand and prescale (for the final (x1+x2)*0.5 norm). f16 output.
// ---------------------------------------------------------------------------
__global__ void k_layernorm(const float* __restrict__ a, const float* __restrict__ b,
                            float prescale, const float* __restrict__ g,
                            const float* __restrict__ be, _Float16* __restrict__ out) {
  int row = blockIdx.x * 4 + threadIdx.y;
  int lane = threadIdx.x;
  const float* pa = a + (size_t)row * DMODEL;
  const float* pb = b ? (b + (size_t)row * DMODEL) : nullptr;
  float xv[4];
  float s = 0.f, s2 = 0.f;
  for (int t = 0; t < 4; ++t) {
    int i = lane + t * 32;
    float v = pa[i];
    if (pb) v += pb[i];
    v *= prescale;
    xv[t] = v;
    s += v; s2 += v * v;
  }
  for (int m = 16; m >= 1; m >>= 1) {
    s  += __shfl_xor(s, m, 32);
    s2 += __shfl_xor(s2, m, 32);
  }
  float mean = s * (1.f / DMODEL);
  float var  = s2 * (1.f / DMODEL) - mean * mean;
  float rs = rsqrtf(var + 1e-5f);
  _Float16* po = out + (size_t)row * DMODEL;
  for (int t = 0; t < 4; ++t) {
    int i = lane + t * 32;
    po[i] = (_Float16)((xv[t] - mean) * rs * g[i] + be[i]);
  }
}

// ---------------------------------------------------------------------------
// WMMA GEMM: C[M,N] = A[M,K] (f16, row-major) x Bt[Npad,K] (f16, pre-transposed)
// One wave computes a 32x64 tile (8 accumulators): each A fragment reused 4x,
// each B fragment 2x. The wave's tile index is forced into an SGPR with
// readfirstlane so all tile guards become scalar branches: EXEC stays all-1
// around every WMMA (ISA requirement) and the K-loop pipelines cleanly.
// flags: ADD -> accumulate into Cf (residual), GELU -> exact gelu epilogue.
// ---------------------------------------------------------------------------
__global__ void k_gemm(const _Float16* __restrict__ A, const _Float16* __restrict__ Bt,
                       const float* __restrict__ bias, float* __restrict__ Cf,
                       _Float16* __restrict__ Ch, int M, int N, int K, int Npad,
                       int flags) {
  int groupsN = (Npad + 63) >> 6;
  int tiles   = (M >> 5) * groupsN;
  // threadIdx.y is uniform within a wave (blockDim.x == 32); make it scalar.
  int tile = __builtin_amdgcn_readfirstlane(blockIdx.x * 4 + threadIdx.y);
  if (tile >= tiles) return;                 // scalar branch
  int tm  = (tile / groupsN) << 5;
  int tn0 = (tile % groupsN) << 6;
  int nsub = (Npad - tn0) >> 4;
  if (nsub > 4) nsub = 4;                    // scalar (tn0, Npad scalar)
  int lane = threadIdx.x;
  int li = lane & 15, hi = lane >> 4;
  // A fragment layout (16x32 f16): lanes 0-15 hold K 0-15, lanes 16-31
  // hold K 16-31 -> two v8h loads + shuffle.
  const _Float16* arow0 = A + (size_t)(tm + li) * K + hi * 16;
  const _Float16* arow1 = arow0 + (size_t)16 * K;
  const _Float16* bbase = Bt + (size_t)(tn0 + li) * K + hi * 16;
  v8f zacc = {};
  v8f acc[8];
  #pragma unroll
  for (int t = 0; t < 8; ++t) acc[t] = zacc;

  if (nsub == 4) {
    // fast path: full 32x64 tile, no guards in the K loop
    for (int k0 = 0; k0 < K; k0 += 32) {
      v8h a0lo = *(const v8h*)(arow0 + k0);
      v8h a0hi = *(const v8h*)(arow0 + k0 + 8);
      v8h a1lo = *(const v8h*)(arow1 + k0);
      v8h a1hi = *(const v8h*)(arow1 + k0 + 8);
      v16h a0 = __builtin_shufflevector(a0lo, a0hi,
                                        0,1,2,3,4,5,6,7,8,9,10,11,12,13,14,15);
      v16h a1 = __builtin_shufflevector(a1lo, a1hi,
                                        0,1,2,3,4,5,6,7,8,9,10,11,12,13,14,15);
      __builtin_prefetch(arow0 + k0 + 64, 0, 0);   // global_prefetch_b8
      __builtin_prefetch(arow1 + k0 + 64, 0, 0);
      #pragma unroll
      for (int sub = 0; sub < 4; ++sub) {
        v16h bm = *(const v16h*)(bbase + (size_t)(sub << 4) * K + k0);
        acc[sub * 2]     = wmma16(a0, bm, acc[sub * 2]);
        acc[sub * 2 + 1] = wmma16(a1, bm, acc[sub * 2 + 1]);
      }
    }
  } else {
    // ragged tail (head GEMMs): scalar per-subtile guard, EXEC untouched
    for (int k0 = 0; k0 < K; k0 += 32) {
      v8h a0lo = *(const v8h*)(arow0 + k0);
      v8h a0hi = *(const v8h*)(arow0 + k0 + 8);
      v8h a1lo = *(const v8h*)(arow1 + k0);
      v8h a1hi = *(const v8h*)(arow1 + k0 + 8);
      v16h a0 = __builtin_shufflevector(a0lo, a0hi,
                                        0,1,2,3,4,5,6,7,8,9,10,11,12,13,14,15);
      v16h a1 = __builtin_shufflevector(a1lo, a1hi,
                                        0,1,2,3,4,5,6,7,8,9,10,11,12,13,14,15);
      #pragma unroll
      for (int sub = 0; sub < 4; ++sub) {
        if (sub < nsub) {                          // scalar condition
          v16h bm = *(const v16h*)(bbase + (size_t)(sub << 4) * K + k0);
          acc[sub * 2]     = wmma16(a0, bm, acc[sub * 2]);
          acc[sub * 2 + 1] = wmma16(a1, bm, acc[sub * 2 + 1]);
        }
      }
    }
  }

  #pragma unroll
  for (int sub = 0; sub < 4; ++sub) {
    if (sub >= nsub) continue;                     // scalar
    int n = tn0 + (sub << 4) + li;
    if (n >= N) continue;                          // per-lane, after all WMMAs
    float bv = bias ? bias[n] : 0.f;
    #pragma unroll
    for (int half = 0; half < 2; ++half) {
      v8f acw = acc[sub * 2 + half];
      for (int j = 0; j < 8; ++j) {
        int m = tm + half * 16 + hi * 8 + j;       // C: lanes 0-15 M=j, 16-31 M=8+j
        float val = acw[j] + bv;
        if (flags & FLAG_GELU) val = gelu_exact(val);
        if (Cf) {
          if (flags & FLAG_ADD) Cf[(size_t)m * N + n] += val;
          else                  Cf[(size_t)m * N + n]  = val;
        }
        if (Ch) Ch[(size_t)m * N + n] = (_Float16)val;
      }
    }
  }
}

// ---------------------------------------------------------------------------
// Split projections into per-head [BH,S,16] f16 buffers; kn = unit-norm qk.
// ---------------------------------------------------------------------------
__global__ void k_split(const float* __restrict__ tq, const float* __restrict__ tv,
                        _Float16* __restrict__ qk, _Float16* __restrict__ kn,
                        _Float16* __restrict__ v) {
  int i = blockIdx.x * blockDim.x + threadIdx.x;
  if (i >= NBH * SEQ) return;
  int bh = i / SEQ, s = i % SEQ;
  int b = bh >> 3, h = bh & 7;
  const float* q  = tq + ((size_t)(b * SEQ + s)) * DMODEL + h * DHEAD;
  const float* vv = tv + ((size_t)(b * SEQ + s)) * DMODEL + h * DHEAD;
  float ss = 0.f, tmp[DHEAD];
  for (int d = 0; d < DHEAD; ++d) { tmp[d] = q[d]; ss += tmp[d] * tmp[d]; }
  float rn = rsqrtf(ss + 1e-9f);
  _Float16* pq = qk + (size_t)i * DHEAD;
  _Float16* pk = kn + (size_t)i * DHEAD;
  _Float16* pv = v  + (size_t)i * DHEAD;
  for (int d = 0; d < DHEAD; ++d) {
    pq[d] = (_Float16)tmp[d];
    pk[d] = (_Float16)(tmp[d] * rn);
    pv[d] = (_Float16)vv[d];
  }
}

// ---------------------------------------------------------------------------
// LSH bucket hashing: rotate qk by rot[d][r][n], argmax over [rot,-rot] (40),
// key = (arg + r*40)*SEQ + s (unique per row). Pad slots get INT_MAX.
// ---------------------------------------------------------------------------
__global__ void k_buckets(const _Float16* __restrict__ qk, const float* __restrict__ rot,
                          int* __restrict__ keyarr) {
  int i = blockIdx.x * blockDim.x + threadIdx.x;
  if (i >= NBH * RSPAD) return;
  int bh = i >> 12, slot = i & (RSPAD - 1);
  int key = 0x7FFFFFFF;
  if (slot < RSLOTS) {
    int r = slot / SEQ, s = slot % SEQ;
    float qf[DHEAD];
    const _Float16* pq = qk + ((size_t)bh * SEQ + s) * DHEAD;
    for (int d = 0; d < DHEAD; ++d) qf[d] = (float)pq[d];
    float best = -3.0e38f; int arg = 0;
    for (int nn = 0; nn < 20; ++nn) {
      float acc = 0.f;
      for (int d = 0; d < DHEAD; ++d)
        acc += qf[d] * rot[d * (NROUND * 20) + r * 20 + nn];
      if (acc > best)  { best = acc;  arg = nn; }
      if (-acc > best) { best = -acc; arg = nn + 20; }
    }
    key = (arg + r * 40) * SEQ + s;
  }
  keyarr[i] = key;
}

// ---------------------------------------------------------------------------
// Per-(batch*head) bitonic sort of 4096 (key,idx) pairs in LDS.
// Produces st = sorted token position, undo = inverse permutation.
// ---------------------------------------------------------------------------
__global__ void k_sort(const int* __restrict__ keyarr, int* __restrict__ st,
                       int* __restrict__ undo) {
  __shared__ int sk[RSPAD];
  __shared__ int si[RSPAD];
  int bh = blockIdx.x;
  int tid = threadIdx.x;
  for (int i = tid; i < RSPAD; i += 1024) {
    sk[i] = keyarr[(size_t)bh * RSPAD + i];
    si[i] = i;
  }
  for (int k2 = 2; k2 <= RSPAD; k2 <<= 1) {
    for (int j = k2 >> 1; j > 0; j >>= 1) {
      __syncthreads();
      for (int i = tid; i < RSPAD; i += 1024) {
        int ixj = i ^ j;
        if (ixj > i) {
          bool up = ((i & k2) == 0);
          int ki = sk[i], kj = sk[ixj];
          if ((ki > kj) == up) {
            sk[i] = kj; sk[ixj] = ki;
            int t = si[i]; si[i] = si[ixj]; si[ixj] = t;
          }
        }
      }
    }
  }
  __syncthreads();
  for (int i = tid; i < RSLOTS; i += 1024) {
    int idx = si[i];
    st[(size_t)bh * RSLOTS + i] = idx % SEQ;
    undo[(size_t)bh * RSLOTS + idx] = i;
  }
}

// ---------------------------------------------------------------------------
// Chunked attention: per (chunk, bh) block of 4 waves.
//  dots (25x50, K=16 pad 32) via 8 WMMA tiles; masks -> LDS; row softmax;
//  attn @ bv (25x16, K=50 pad 64) via 2x2 WMMA. Outputs bo_sorted + lse.
// Wave phase index forced scalar via readfirstlane (EXEC all-1 around WMMA).
// ---------------------------------------------------------------------------
__global__ void k_attend(const _Float16* __restrict__ qk, const _Float16* __restrict__ kn,
                         const _Float16* __restrict__ v, const int* __restrict__ st,
                         const int* __restrict__ keep, float* __restrict__ bo_s,
                         float* __restrict__ lse_s) {
  int n  = blockIdx.x;   // chunk
  int bh = blockIdx.y;
  int b  = bh >> 3;
  int tid = threadIdx.y * 32 + threadIdx.x;

  __shared__ int s_tq[32];
  __shared__ int s_tkv[64];
  __shared__ unsigned char s_mq[32];
  __shared__ unsigned char s_mkv[64];
  __shared__ __align__(32) float    sdots[32][64];
  __shared__ __align__(32) _Float16 sattn[32][64];
  __shared__ __align__(32) _Float16 sbvT[16][64];

  if (tid < 32) {
    int t = 0, mq = 0;
    if (tid < CHUNK) {
      t = st[(size_t)bh * RSLOTS + n * CHUNK + tid];
      mq = keep[b * SEQ + t];
    }
    s_tq[tid] = (tid < CHUNK) ? t : -1;
    s_mq[tid] = (unsigned char)mq;
  }
  if (tid < 64) {
    int t = 0, mk = 0;
    if (tid < CHUNK) {
      t = st[(size_t)bh * RSLOTS + n * CHUNK + tid];
      mk = keep[b * SEQ + t];
    } else if (tid < 2 * CHUNK) {
      int pn = (n == 0) ? (NCHUNK - 1) : (n - 1);      // look-back: previous chunk
      t = st[(size_t)bh * RSLOTS + pn * CHUNK + (tid - CHUNK)];
      mk = keep[b * SEQ + t];
    }
    s_tkv[tid] = (tid < 2 * CHUNK) ? t : -1;
    s_mkv[tid] = (unsigned char)mk;
  }
  __syncthreads();

  // stage V^T tile in LDS (zeros for padded K rows)
  if (tid < 64) {
    if (tid < 2 * CHUNK) {
      const _Float16* vr = v + ((size_t)bh * SEQ + s_tkv[tid]) * DHEAD;
      for (int d = 0; d < DHEAD; ++d) sbvT[d][tid] = vr[d];
    } else {
      for (int d = 0; d < DHEAD; ++d) sbvT[d][tid] = (_Float16)0.f;
    }
  }
  __syncthreads();

  int lane = threadIdx.x;
  int li = lane & 15, hi = lane >> 4;
  int wy = __builtin_amdgcn_readfirstlane(threadIdx.y);  // scalar wave id
  v8h zed = {};

  // --- dots = (q . k) * D^-1/2, 8 tiles (2 M-tiles x 4 N-tiles), K=16 pad 32
  #pragma unroll
  for (int half2 = 0; half2 < 2; ++half2) {
    int tile = wy + half2 * 4;                 // scalar
    int mt = tile >> 2, nt = tile & 3;
    v16h a = {};
    int mrow = mt * 16 + li;
    if (mrow < CHUNK && hi == 0) {
      const _Float16* ar = qk + ((size_t)bh * SEQ + s_tq[mrow]) * DHEAD;
      a = *(const v16h*)ar;                   // K 0-15 (lanes 0-15); lanes 16-31 pad -> 0
    }
    v16h bm = {};
    int kc = nt * 16 + li;
    if (hi == 0 && kc < 2 * CHUNK) {          // lanes 16-31 are K=16..31 pad -> 0
      bm = *(const v16h*)(kn + ((size_t)bh * SEQ + s_tkv[kc]) * DHEAD);
    }
    v8f acc = {};
    acc = wmma16(a, bm, acc);
    for (int j = 0; j < 8; ++j) {
      int mr = mt * 16 + hi * 8 + j;
      int nc = nt * 16 + li;
      float dv = NEGV;
      if (mr < CHUNK && nc < 2 * CHUNK) {
        dv = acc[j] * 0.25f;                  // D^-0.5, D=16
        if (!(s_mq[mr] && s_mkv[nc])) dv = NEGV;   // pad mask
        if (s_tq[mr] <  s_tkv[nc]) dv = NEGV;      // causal
        if (s_tq[mr] == s_tkv[nc]) dv = SELFV;     // discourage self
      }
      sdots[mr][nc] = dv;
    }
  }
  __syncthreads();

  // --- row softmax + lse
  if (tid < 32) {
    if (tid < CHUNK) {
      float mx = -3.0e38f;
      for (int k2 = 0; k2 < 2 * CHUNK; ++k2) mx = fmaxf(mx, sdots[tid][k2]);
      float ssum = 0.f;
      for (int k2 = 0; k2 < 2 * CHUNK; ++k2) ssum += expf(sdots[tid][k2] - mx);
      float l_ = logf(ssum) + mx;
      lse_s[(size_t)bh * RSLOTS + n * CHUNK + tid] = l_;
      for (int k2 = 0; k2 < 64; ++k2)
        sattn[tid][k2] = (_Float16)((k2 < 2 * CHUNK) ? expf(sdots[tid][k2] - l_) : 0.f);
    } else {
      for (int k2 = 0; k2 < 64; ++k2) sattn[tid][k2] = (_Float16)0.f;
    }
  }
  __syncthreads();

  // --- out = attn @ bv : 2 M-tiles x 1 N-tile, K=64 (2 wmma steps)
  if (wy < 2) {                                // scalar branch
    v8f acc = {};
    for (int k0 = 0; k0 < 64; k0 += 32) {
      v8h alo = *(const v8h*)(&sattn[wy * 16 + li][k0 + hi * 16]);
      v8h ahi = *(const v8h*)(&sattn[wy * 16 + li][k0 + hi * 16 + 8]);
      v16h a = __builtin_shufflevector(alo, ahi,
                                       0,1,2,3,4,5,6,7,8,9,10,11,12,13,14,15);
      v16h bm = *(const v16h*)(&sbvT[li][k0 + hi * 16]);
      acc = wmma16(a, bm, acc);
    }
    for (int j = 0; j < 8; ++j) {
      int mr = wy * 16 + hi * 8 + j;
      if (mr < CHUNK)
        bo_s[((size_t)bh * RSLOTS + n * CHUNK + mr) * DHEAD + li] = acc[j];
    }
  }
}

// ---------------------------------------------------------------------------
// Undo sort and combine hash rounds by softmax of their logsumexps.
// Writes heads merged back to [B,S,128] f16 (input of the Wo GEMM).
// ---------------------------------------------------------------------------
__global__ void k_unsort(const float* __restrict__ bo_s, const float* __restrict__ lse_s,
                         const int* __restrict__ undo, _Float16* __restrict__ attn16) {
  int i = blockIdx.x * blockDim.x + threadIdx.x;
  if (i >= NBH * SEQ) return;
  int bh = i / SEQ, s = i % SEQ;
  int p[NROUND]; float l[NROUND];
  float mx = -3.0e38f;
  for (int r = 0; r < NROUND; ++r) {
    p[r] = undo[(size_t)bh * RSLOTS + r * SEQ + s];
    l[r] = lse_s[(size_t)bh * RSLOTS + p[r]];
    mx = fmaxf(mx, l[r]);
  }
  float wsum = 0.f;
  for (int r = 0; r < NROUND; ++r) { l[r] = expf(l[r] - mx); wsum += l[r]; }
  float inv = 1.f / wsum;
  int b = bh >> 3, h = bh & 7;
  _Float16* po = attn16 + ((size_t)(b * SEQ + s)) * DMODEL + h * DHEAD;
  for (int d = 0; d < DHEAD; ++d) {
    float acc = 0.f;
    for (int r = 0; r < NROUND; ++r)
      acc += l[r] * bo_s[((size_t)bh * RSLOTS + p[r]) * DHEAD + d];
    po[d] = (_Float16)(acc * inv);
  }
}

// ---------------------------------------------------------------------------
// Product-key memory: per (token, pkm-head) thread. top16 x top16 -> top16,
// softmax, gather values, atomic-accumulate into x2 (sums over heads).
// ---------------------------------------------------------------------------
__global__ void k_pkm(const float* __restrict__ q, const float* __restrict__ keys,
                      const float* __restrict__ values, float* __restrict__ x2) {
  int t = blockIdx.x * blockDim.x + threadIdx.x;
  if (t >= NTOK * 4) return;
  int tok = t >> 2, h = t & 3;
  const float* qp = q + (size_t)tok * 512 + h * 128;  // [2][64]
  float sx[16], sy[16]; int ix[16], iy[16];
  for (int k = 0; k < 16; ++k) { sx[k] = -3.0e38f; sy[k] = -3.0e38f; ix[k] = 0; iy[k] = 0; }
  for (int nn = 0; nn < 128; ++nn) {
    const float* kb = keys + (size_t)(h * 128 + nn) * 128;  // [p][d]
    float d0 = 0.f, d1 = 0.f;
    for (int d = 0; d < 64; ++d) { d0 += qp[d] * kb[d]; d1 += qp[64 + d] * kb[64 + d]; }
    if (d0 > sx[15]) {
      int j = 15; while (j > 0 && sx[j-1] < d0) { sx[j] = sx[j-1]; ix[j] = ix[j-1]; --j; }
      sx[j] = d0; ix[j] = nn;
    }
    if (d1 > sy[15]) {
      int j = 15; while (j > 0 && sy[j-1] < d1) { sy[j] = sy[j-1]; iy[j] = iy[j-1]; --j; }
      sy[j] = d1; iy[j] = nn;
    }
  }
  float fs[16]; int fk[16];
  for (int k = 0; k < 16; ++k) { fs[k] = -3.0e38f; fk[k] = 0; }
  for (int a = 0; a < 16; ++a)
    for (int b2 = 0; b2 < 16; ++b2) {
      float vsum = sx[a] + sy[b2];
      if (vsum > fs[15]) {
        int id = ix[a] * 128 + iy[b2];
        int j = 15; while (j > 0 && fs[j-1] < vsum) { fs[j] = fs[j-1]; fk[j] = fk[j-1]; --j; }
        fs[j] = vsum; fk[j] = id;
      }
    }
  float mx = fs[0], ssum = 0.f, w[16];
  for (int k = 0; k < 16; ++k) { w[k] = expf(fs[k] - mx); ssum += w[k]; }
  float inv = 1.f / ssum;
  for (int d = 0; d < 128; ++d) {
    float acc = 0.f;
    for (int k = 0; k < 16; ++k) acc += w[k] * values[(size_t)fk[k] * 128 + d];
    atomicAdd(&x2[(size_t)tok * 128 + d], acc * inv);
  }
}

// ---------------------------------------------------------------------------
// Host orchestration
// ---------------------------------------------------------------------------
extern "C" void kernel_launch(void* const* d_in, const int* in_sizes, int n_in,
                              void* d_out, int out_size, void* d_ws, size_t ws_size,
                              hipStream_t stream) {
  (void)in_sizes; (void)n_in; (void)out_size; (void)ws_size;
  const int*   x_mcc    = (const int*)d_in[0];
  const int*   x_amt    = (const int*)d_in[1];
  const int*   x_hour   = (const int*)d_in[2];
  const int*   x_wday   = (const int*)d_in[3];
  const float* emb_mcc  = (const float*)d_in[4];
  const float* emb_amt  = (const float*)d_in[5];
  const float* emb_hour = (const float*)d_in[6];
  const float* emb_wday = (const float*)d_in[7];
  const float* rot      = (const float*)d_in[8];
  const float* ln1_g    = (const float*)d_in[9];
  const float* ln1_b    = (const float*)d_in[10];
  const float* ln2_g    = (const float*)d_in[11];
  const float* ln2_b    = (const float*)d_in[12];
  const float* Wqk      = (const float*)d_in[13];
  const float* Wv       = (const float*)d_in[14];
  const float* Wo       = (const float*)d_in[15];
  const float* bo       = (const float*)d_in[16];
  const float* ff_w1    = (const float*)d_in[17];
  const float* ff_b1    = (const float*)d_in[18];
  const float* ff_w2    = (const float*)d_in[19];
  const float* ff_b2    = (const float*)d_in[20];
  const float* pkm_wq   = (const float*)d_in[21];
  const float* pkm_keys = (const float*)d_in[22];
  const float* pkm_vals = (const float*)d_in[23];
  const float* norm_g   = (const float*)d_in[24];
  const float* norm_b   = (const float*)d_in[25];
  const float* hw1      = (const float*)d_in[26];
  const float* hb1      = (const float*)d_in[27];
  const float* hw2[4]   = {(const float*)d_in[28], (const float*)d_in[30],
                           (const float*)d_in[32], (const float*)d_in[34]};
  const float* hb2[4]   = {(const float*)d_in[29], (const float*)d_in[31],
                           (const float*)d_in[33], (const float*)d_in[35]};
  float* out = (float*)d_out;

  // -- bump allocator over d_ws
  char* wsp = (char*)d_ws;
  auto alloc = [&](size_t bytes) -> char* {
    char* p = wsp; wsp += (bytes + 255) & ~(size_t)255; return p;
  };
  float*    x1     = (float*)   alloc((size_t)NTOK * DMODEL * 4);
  float*    x2     = (float*)   alloc((size_t)NTOK * DMODEL * 4);
  float*    tmpQ   = (float*)   alloc((size_t)NTOK * DMODEL * 4);
  float*    tmpV   = (float*)   alloc((size_t)NTOK * DMODEL * 4);
  float*    tmpF   = (float*)   alloc((size_t)NTOK * 512 * 4);
  float*    lse_s  = (float*)   alloc((size_t)NBH * RSLOTS * 4);
  float*    bo_s   = (float*)   alloc((size_t)NBH * RSLOTS * DHEAD * 4);
  _Float16* h16    = (_Float16*)alloc((size_t)NTOK * DMODEL * 2);
  _Float16* ff1h   = (_Float16*)alloc((size_t)NTOK * 512 * 2);
  _Float16* attn16 = (_Float16*)alloc((size_t)NTOK * DMODEL * 2);
  _Float16* y16    = (_Float16*)alloc((size_t)NTOK * DMODEL * 2);
  _Float16* hh16   = (_Float16*)alloc((size_t)NTOK * DMODEL * 2);
  _Float16* qk16   = (_Float16*)alloc((size_t)NBH * SEQ * DHEAD * 2);
  _Float16* kn16   = (_Float16*)alloc((size_t)NBH * SEQ * DHEAD * 2);
  _Float16* v16    = (_Float16*)alloc((size_t)NBH * SEQ * DHEAD * 2);
  int*      keep   = (int*)     alloc((size_t)NTOK * 4);
  int*      keyarr = (int*)     alloc((size_t)NBH * RSPAD * 4);
  int*      st     = (int*)     alloc((size_t)NBH * RSLOTS * 4);
  int*      undo   = (int*)     alloc((size_t)NBH * RSLOTS * 4);
  _Float16* wqkT   = (_Float16*)alloc((size_t)8 * 128 * 128 * 2);
  _Float16* wvT    = (_Float16*)alloc((size_t)8 * 128 * 128 * 2);
  _Float16* woT    = (_Float16*)alloc((size_t)8 * 128 * 128 * 2);
  _Float16* ffw1T  = (_Float16*)alloc((size_t)6 * 512 * 128 * 2);
  _Float16* ffw2T  = (_Float16*)alloc((size_t)6 * 128 * 512 * 2);
  _Float16* pkmqT  = (_Float16*)alloc((size_t)2 * 512 * 128 * 2);
  _Float16* hw1T   = (_Float16*)alloc((size_t)4 * 128 * 128 * 2);
  _Float16* hw2T   = (_Float16*)alloc((size_t)512 * 128 * 2);

  auto conv = [&](const float* src, _Float16* dst, int K, int N, int Npad) {
    int total = Npad * K;
    k_w2h<<<(total + 255) / 256, 256, 0, stream>>>(src, dst, K, N, Npad);
  };
  auto gemm = [&](const _Float16* A, const _Float16* Bt, const float* bias,
                  float* Cf, _Float16* Ch, int M, int N, int K, int Npad, int flags) {
    int tiles = (M / 32) * ((Npad + 63) / 64);
    k_gemm<<<(tiles + 3) / 4, dim3(32, 4), 0, stream>>>(A, Bt, bias, Cf, Ch,
                                                        M, N, K, Npad, flags);
  };

  // --- weight conversion / transpose (f32 -> f16 Bt) ---
  for (int l = 0; l < 8; ++l) {
    conv(Wqk + (size_t)l * 16384, wqkT + (size_t)l * 16384, 128, 128, 128);
    conv(Wv  + (size_t)l * 16384, wvT  + (size_t)l * 16384, 128, 128, 128);
    conv(Wo  + (size_t)l * 16384, woT  + (size_t)l * 16384, 128, 128, 128);
  }
  for (int f = 0; f < 6; ++f) {
    conv(ff_w1 + (size_t)f * 128 * 512, ffw1T + (size_t)f * 512 * 128, 128, 512, 512);
    conv(ff_w2 + (size_t)f * 512 * 128, ffw2T + (size_t)f * 128 * 512, 512, 128, 128);
  }
  for (int p = 0; p < 2; ++p)
    conv(pkm_wq + (size_t)p * 128 * 512, pkmqT + (size_t)p * 512 * 128, 128, 512, 512);
  for (int i = 0; i < 4; ++i)
    conv(hw1 + (size_t)i * 16384, hw1T + (size_t)i * 16384, 128, 128, 128);
  int Nh[4]  = {350, 100, 25, 8};
  int Nph[4] = {352, 112, 32, 16};
  size_t hw2off[4]; size_t hoff = 0;
  for (int i = 0; i < 4; ++i) {
    hw2off[i] = hoff;
    conv(hw2[i], hw2T + hoff, 128, Nh[i], Nph[i]);
    hoff += (size_t)Nph[i] * 128;
  }

  // --- embeddings + positional encoding ---
  k_embed<<<(NTOK + 255) / 256, 256, 0, stream>>>(x_mcc, x_amt, x_hour, x_wday,
                                                  emb_mcc, emb_amt, emb_hour, emb_wday,
                                                  x1, x2, keep);

  // --- reversible layers ---
  int ffi = 0, pki = 0;
  for (int l = 0; l < 8; ++l) {
    // h = LN(x2)
    k_layernorm<<<NTOK / 4, dim3(32, 4), 0, stream>>>(x2, nullptr, 1.f,
                                                      ln1_g + l * 128, ln1_b + l * 128, h16);
    // qk, v projections
    gemm(h16, wqkT + (size_t)l * 16384, nullptr, tmpQ, nullptr, NTOK, 128, 128, 128, 0);
    gemm(h16, wvT  + (size_t)l * 16384, nullptr, tmpV, nullptr, NTOK, 128, 128, 128, 0);
    k_split<<<(NBH * SEQ + 255) / 256, 256, 0, stream>>>(tmpQ, tmpV, qk16, kn16, v16);
    // hash, sort, chunked attend, unsort+combine
    k_buckets<<<(NBH * RSPAD + 255) / 256, 256, 0, stream>>>(qk16, rot, keyarr);
    k_sort<<<NBH, 1024, 0, stream>>>(keyarr, st, undo);
    k_attend<<<dim3(NCHUNK, NBH), dim3(32, 4), 0, stream>>>(qk16, kn16, v16, st, keep,
                                                            bo_s, lse_s);
    k_unsort<<<(NBH * SEQ + 255) / 256, 256, 0, stream>>>(bo_s, lse_s, undo, attn16);
    // x1 += attn @ Wo + bo
    gemm(attn16, woT + (size_t)l * 16384, bo + l * 128, x1, nullptr,
         NTOK, 128, 128, 128, FLAG_ADD);
    // h2 = LN(x1)
    k_layernorm<<<NTOK / 4, dim3(32, 4), 0, stream>>>(x1, nullptr, 1.f,
                                                      ln2_g + l * 128, ln2_b + l * 128, h16);
    if (l == 3 || l == 6) {
      gemm(h16, pkmqT + (size_t)pki * 512 * 128, nullptr, tmpF, nullptr,
           NTOK, 512, 128, 512, 0);
      k_pkm<<<(NTOK * 4 + 63) / 64, 64, 0, stream>>>(tmpF,
           pkm_keys + (size_t)pki * 4 * 128 * 2 * 64,
           pkm_vals + (size_t)pki * 16384 * 128, x2);
      ++pki;
    } else {
      gemm(h16, ffw1T + (size_t)ffi * 512 * 128, ff_b1 + ffi * 512, nullptr, ff1h,
           NTOK, 512, 128, 512, FLAG_GELU);
      gemm(ff1h, ffw2T + (size_t)ffi * 128 * 512, ff_b2 + ffi * 128, x2, nullptr,
           NTOK, 128, 512, 128, FLAG_ADD);
      ++ffi;
    }
  }

  // --- final norm + heads ---
  k_layernorm<<<NTOK / 4, dim3(32, 4), 0, stream>>>(x1, x2, 0.5f, norm_g, norm_b, y16);
  size_t ooff = 0;
  for (int i = 0; i < 4; ++i) {
    gemm(y16, hw1T + (size_t)i * 16384, hb1 + i * 128, nullptr, hh16,
         NTOK, 128, 128, 128, FLAG_GELU);
    gemm(hh16, hw2T + hw2off[i], hb2[i], out + ooff, nullptr,
         NTOK, Nh[i], 128, Nph[i], 0);
    ooff += (size_t)NTOK * Nh[i];
  }
}
